// GraphConvolution_27711128994144
// MI455X (gfx1250) — matrix operand order repaired and checked by reading, not compiled, not run on hardware
//
#include <hip/hip_runtime.h>
#include <hip/hip_bf16.h>

#define NUM_R 5
#define N_U   4096
#define N_V   8192
#define D_IN  512
#define D_H   256

#define S_U 4   // k-splits for z_u (over N_V)  -> grid 64x4  = 256 blocks
#define S_V 2   // k-splits for z_v (over N_U)  -> grid 128x2 = 256 blocks

typedef __attribute__((ext_vector_type(8)))  _Float16 v8h;
typedef __attribute__((ext_vector_type(16))) _Float16 v16h;
typedef __attribute__((ext_vector_type(8)))  float    v8f;

#define BK  32
#define LDK 40   // BK + 8 halves pad: rows stay 16B-aligned, 80B stride spreads banks

// ---- WMMA fragment gathers (CDNA5 16-bit A/B VGPR layouts, wave32) ----
// A 16x32: lanes 0-15 -> M=lane, K chunks {0..7},{16..23}; lanes 16-31 -> {8..15},{24..31}
__device__ __forceinline__ v16h frag_a(const _Float16* row, int laneHi) {
  v8h lo = *(const v8h*)(row + 8 * laneHi);
  v8h hi = *(const v8h*)(row + 16 + 8 * laneHi);
  return __builtin_shufflevector(lo, hi, 0,1,2,3,4,5,6,7,8,9,10,11,12,13,14,15);
}
// B 32x16 stored [N][K] in LDS: lanes 0-15 -> N=lane, K chunks {0..7},{16..23}; lanes 16-31 -> {8..15},{24..31}
__device__ __forceinline__ v16h frag_b(const _Float16* row, int laneHi) {
  v8h lo = *(const v8h*)(row + 8 * laneHi);
  v8h hi = *(const v8h*)(row + 16 + 8 * laneHi);
  return __builtin_shufflevector(lo, hi, 0,1,2,3,4,5,6,7,8,9,10,11,12,13,14,15);
}

__device__ __forceinline__ v8f wmma_f16(v16h a, v16h b, v8f c) {
  return __builtin_amdgcn_wmma_f32_16x16x32_f16(false, a, false, b, (short)0, c, false, false);
}

// ---------------- prep kernels ----------------

__global__ void k_cumsum_w(const float* __restrict__ w, _Float16* __restrict__ wc) {
  int idx = blockIdx.x * blockDim.x + threadIdx.x;
  if (idx >= D_IN * D_H) return;
  float acc = 0.f;
#pragma unroll
  for (int r = 0; r < NUM_R; ++r) {
    acc += w[r * D_IN * D_H + idx];
    wc[r * D_IN * D_H + idx] = (_Float16)acc;
  }
}

__global__ void k_colsum(const float* __restrict__ sup, float* __restrict__ rinv) {
  int r = blockIdx.y;
  int m = blockIdx.x * 256 + threadIdx.x;
  const float* p = sup + (size_t)r * N_U * N_V + m;
  float s = 0.f;
  for (int n = 0; n < N_U; n += 4) {
    s += p[(size_t)n * N_V] + p[(size_t)(n + 1) * N_V] +
         p[(size_t)(n + 2) * N_V] + p[(size_t)(n + 3) * N_V];
  }
  rinv[r * N_V + m] = (s > 0.f) ? rsqrtf(s) : 0.f;
}

__global__ void k_rowsum(const float* __restrict__ sup, float* __restrict__ cinv) {
  int r = blockIdx.y, n = blockIdx.x, t = threadIdx.x;
  const float* p = sup + ((size_t)r * N_U + n) * N_V;
  float s = 0.f;
  for (int j = t; j < N_V; j += 256) s += p[j];
  __shared__ float red[256];
  red[t] = s;
  __syncthreads();
  for (int w = 128; w > 0; w >>= 1) {
    if (t < w) red[t] += red[t + w];
    __syncthreads();
  }
  if (t == 0) {
    float v = red[0];
    cinv[r * N_U + n] = (v > 0.f) ? rsqrtf(v) : 0.f;
  }
}

// ------------- GEMM1: tmp[r] = (feat @ w_cum[r]) * scale_row, stored f16 -------------

__global__ void k_gemm_feat(const float* __restrict__ feat, int nrows,
                            const _Float16* __restrict__ wc,
                            const float* __restrict__ scale,   // [NUM_R * nrows]
                            _Float16* __restrict__ out) {      // [NUM_R][nrows][D_H]
  const int n0 = blockIdx.x * 64, h0 = blockIdx.y * 64, r = blockIdx.z;
  const int t = threadIdx.x, lane = t & 31, wave = t >> 5;
  const int laneHi = lane >> 4, lanelo = lane & 15;
  const int msub = wave & 3, hsub = (wave >> 2) * 2;
  __shared__ __align__(16) _Float16 As[64][LDK];
  __shared__ __align__(16) _Float16 Bs[64][LDK];
  v8f acc0 = {}, acc1 = {};
  const int rowA = t >> 2, c4 = t & 3;
  const int hloc = t & 63, kgrp = t >> 6;

  for (int k0 = 0; k0 < D_IN; k0 += BK) {
    const float* ap = feat + (size_t)(n0 + rowA) * D_IN + k0 + c4 * 8;
    float4 x = *(const float4*)ap;
    float4 y = *(const float4*)(ap + 4);
    v8h av;
    av[0] = (_Float16)x.x; av[1] = (_Float16)x.y; av[2] = (_Float16)x.z; av[3] = (_Float16)x.w;
    av[4] = (_Float16)y.x; av[5] = (_Float16)y.y; av[6] = (_Float16)y.z; av[7] = (_Float16)y.w;
    *(v8h*)&As[rowA][c4 * 8] = av;

    v8h bv;
#pragma unroll
    for (int j = 0; j < 8; ++j)
      bv[j] = wc[(size_t)(r * D_IN + k0 + kgrp * 8 + j) * D_H + h0 + hloc];
    *(v8h*)&Bs[hloc][kgrp * 8] = bv;
    __syncthreads();

    v16h a  = frag_a(&As[msub * 16 + lanelo][0], laneHi);
    v16h b0 = frag_b(&Bs[hsub * 16 + lanelo][0], laneHi);
    v16h b1 = frag_b(&Bs[(hsub + 1) * 16 + lanelo][0], laneHi);
    acc0 = wmma_f16(a, b0, acc0);
    acc1 = wmma_f16(a, b1, acc1);
    __syncthreads();
  }
#pragma unroll
  for (int v = 0; v < 8; ++v) {
    int M = msub * 16 + v + 8 * laneHi;
    int row = n0 + M;
    float sc = scale[r * nrows + row];
    size_t base = ((size_t)r * nrows + row) * D_H + h0;
    out[base + hsub * 16 + lanelo]        = (_Float16)(acc0[v] * sc);
    out[base + (hsub + 1) * 16 + lanelo]  = (_Float16)(acc1[v] * sc);
  }
}

// ------------- GEMM2a: partial_u[s] += (cinv*sup-rows) @ tmpv   (64 x fullH tile, split-K) -------------

__global__ void k_gemm_zu(const float* __restrict__ sup,
                          const int* __restrict__ uidx,
                          const float* __restrict__ cinv,      // [NUM_R*N_U]
                          const _Float16* __restrict__ tmpv,   // pre-scaled by r_inv
                          float* __restrict__ partial) {       // [S_U][N_U][D_H]
  const int n0 = blockIdx.x * 64;
  const int split = blockIdx.y;
  const int kbeg = split * (N_V / S_U), kend = kbeg + (N_V / S_U);
  const int t = threadIdx.x, lane = t & 31, wave = t >> 5;
  const int laneHi = lane >> 4, lanelo = lane & 15;
  const int msub = wave & 3, hhalf = wave >> 2;   // wave owns 8 H subtiles
  __shared__ __align__(16) _Float16 As[64][LDK];
  __shared__ __align__(16) _Float16 Bs[D_H][LDK];
  __shared__ int   nIdx[64];
  __shared__ float nScale[64];
  if (t < 64) nIdx[t] = uidx[n0 + t];
  __syncthreads();

  v8f zero = {};
  v8f acc[8];
#pragma unroll
  for (int i = 0; i < 8; ++i) acc[i] = zero;

  const int rowA = t >> 2, c4 = t & 3;

  for (int r = 0; r < NUM_R; ++r) {
    if (t < 64) nScale[t] = cinv[r * N_U + nIdx[t]];
    // prologue prefetch (global -> regs)
    float4 ax, ay;
    v8h b0, b1, b2, b3;
    {
      const float* ap = sup + ((size_t)r * N_U + nIdx[rowA]) * N_V + kbeg + c4 * 8;
      ax = *(const float4*)ap; ay = *(const float4*)(ap + 4);
      const _Float16* bp = tmpv + ((size_t)r * N_V + kbeg) * D_H + t;
#pragma unroll
      for (int j = 0; j < 8; ++j) b0[j] = bp[(size_t)j * D_H];
#pragma unroll
      for (int j = 0; j < 8; ++j) b1[j] = bp[(size_t)(8 + j) * D_H];
#pragma unroll
      for (int j = 0; j < 8; ++j) b2[j] = bp[(size_t)(16 + j) * D_H];
#pragma unroll
      for (int j = 0; j < 8; ++j) b3[j] = bp[(size_t)(24 + j) * D_H];
    }
    for (int k0 = kbeg; k0 < kend; k0 += BK) {
      __syncthreads();   // prev compute done; nScale visible
      {
        float s = nScale[rowA];
        v8h av;
        av[0] = (_Float16)(ax.x * s); av[1] = (_Float16)(ax.y * s);
        av[2] = (_Float16)(ax.z * s); av[3] = (_Float16)(ax.w * s);
        av[4] = (_Float16)(ay.x * s); av[5] = (_Float16)(ay.y * s);
        av[6] = (_Float16)(ay.z * s); av[7] = (_Float16)(ay.w * s);
        *(v8h*)&As[rowA][c4 * 8] = av;
        *(v8h*)&Bs[t][0]  = b0;
        *(v8h*)&Bs[t][8]  = b1;
        *(v8h*)&Bs[t][16] = b2;
        *(v8h*)&Bs[t][24] = b3;
      }
      const bool more = (k0 + BK) < kend;
      float4 nax, nay;
      v8h nb0, nb1, nb2, nb3;
      if (more) {   // uniform branch: overlap next tile's HBM latency with WMMA
        const float* ap = sup + ((size_t)r * N_U + nIdx[rowA]) * N_V + (k0 + BK) + c4 * 8;
        nax = *(const float4*)ap; nay = *(const float4*)(ap + 4);
        const _Float16* bp = tmpv + ((size_t)r * N_V + (k0 + BK)) * D_H + t;
#pragma unroll
        for (int j = 0; j < 8; ++j) nb0[j] = bp[(size_t)j * D_H];
#pragma unroll
        for (int j = 0; j < 8; ++j) nb1[j] = bp[(size_t)(8 + j) * D_H];
#pragma unroll
        for (int j = 0; j < 8; ++j) nb2[j] = bp[(size_t)(16 + j) * D_H];
#pragma unroll
        for (int j = 0; j < 8; ++j) nb3[j] = bp[(size_t)(24 + j) * D_H];
      }
      __syncthreads();   // staging complete
      v16h a = frag_a(&As[msub * 16 + lanelo][0], laneHi);
#pragma unroll
      for (int i = 0; i < 8; ++i) {
        v16h b = frag_b(&Bs[hhalf * 128 + i * 16 + lanelo][0], laneHi);
        acc[i] = wmma_f16(a, b, acc[i]);
      }
      if (more) { ax = nax; ay = nay; b0 = nb0; b1 = nb1; b2 = nb2; b3 = nb3; }
    }
  }
#pragma unroll
  for (int i = 0; i < 8; ++i) {
    int col = hhalf * 128 + i * 16 + lanelo;
#pragma unroll
    for (int v = 0; v < 8; ++v) {
      int M = msub * 16 + v + 8 * laneHi;
      partial[((size_t)split * N_U + n0 + M) * D_H + col] = acc[i][v];
    }
  }
}

// ------------- GEMM2b: partial_v[s] += (sup^T * rinv-rows) @ tmpu  (transpose via LDS) -------------

__global__ void k_gemm_zv(const float* __restrict__ sup,
                          const int* __restrict__ vidx,
                          const float* __restrict__ rinv,      // [NUM_R*N_V]
                          const _Float16* __restrict__ tmpu,   // pre-scaled by c_inv
                          float* __restrict__ partial) {       // [S_V][N_V][D_H]
  const int m0 = blockIdx.x * 64;
  const int split = blockIdx.y;
  const int kbeg = split * (N_U / S_V), kend = kbeg + (N_U / S_V);
  const int t = threadIdx.x, lane = t & 31, wave = t >> 5;
  const int laneHi = lane >> 4, lanelo = lane & 15;
  const int msub = wave & 3, hhalf = wave >> 2;
  __shared__ __align__(16) _Float16 As[64][LDK];
  __shared__ __align__(16) _Float16 Bs[D_H][LDK];
  __shared__ int   mIdx[64];
  __shared__ float mScale[64];
  if (t < 64) mIdx[t] = vidx[m0 + t];
  __syncthreads();

  v8f zero = {};
  v8f acc[8];
#pragma unroll
  for (int i = 0; i < 8; ++i) acc[i] = zero;

  const int mloc = t & 63, ngrp = t >> 6;   // A: transpose through LDS, coalesced across mloc

  for (int r = 0; r < NUM_R; ++r) {
    if (t < 64) mScale[t] = rinv[r * N_V + mIdx[t]];
    float af[8];
    v8h b0, b1, b2, b3;
    {
      const float* ap = sup + ((size_t)(r * N_U + kbeg + ngrp * 8)) * N_V + mIdx[mloc];
#pragma unroll
      for (int j = 0; j < 8; ++j) af[j] = ap[(size_t)j * N_V];
      const _Float16* bp = tmpu + ((size_t)r * N_U + kbeg) * D_H + t;
#pragma unroll
      for (int j = 0; j < 8; ++j) b0[j] = bp[(size_t)j * D_H];
#pragma unroll
      for (int j = 0; j < 8; ++j) b1[j] = bp[(size_t)(8 + j) * D_H];
#pragma unroll
      for (int j = 0; j < 8; ++j) b2[j] = bp[(size_t)(16 + j) * D_H];
#pragma unroll
      for (int j = 0; j < 8; ++j) b3[j] = bp[(size_t)(24 + j) * D_H];
    }
    for (int k0 = kbeg; k0 < kend; k0 += BK) {
      __syncthreads();
      {
        float s = mScale[mloc];
        v8h av;
#pragma unroll
        for (int j = 0; j < 8; ++j) av[j] = (_Float16)(af[j] * s);
        *(v8h*)&As[mloc][ngrp * 8] = av;
        *(v8h*)&Bs[t][0]  = b0;
        *(v8h*)&Bs[t][8]  = b1;
        *(v8h*)&Bs[t][16] = b2;
        *(v8h*)&Bs[t][24] = b3;
      }
      const bool more = (k0 + BK) < kend;
      float naf[8];
      v8h nb0, nb1, nb2, nb3;
      if (more) {
        const float* ap = sup + ((size_t)(r * N_U + k0 + BK + ngrp * 8)) * N_V + mIdx[mloc];
#pragma unroll
        for (int j = 0; j < 8; ++j) naf[j] = ap[(size_t)j * N_V];
        const _Float16* bp = tmpu + ((size_t)r * N_U + (k0 + BK)) * D_H + t;
#pragma unroll
        for (int j = 0; j < 8; ++j) nb0[j] = bp[(size_t)j * D_H];
#pragma unroll
        for (int j = 0; j < 8; ++j) nb1[j] = bp[(size_t)(8 + j) * D_H];
#pragma unroll
        for (int j = 0; j < 8; ++j) nb2[j] = bp[(size_t)(16 + j) * D_H];
#pragma unroll
        for (int j = 0; j < 8; ++j) nb3[j] = bp[(size_t)(24 + j) * D_H];
      }
      __syncthreads();
      v16h a = frag_a(&As[msub * 16 + lanelo][0], laneHi);
#pragma unroll
      for (int i = 0; i < 8; ++i) {
        v16h b = frag_b(&Bs[hhalf * 128 + i * 16 + lanelo][0], laneHi);
        acc[i] = wmma_f16(a, b, acc[i]);
      }
      if (more) {
#pragma unroll
        for (int j = 0; j < 8; ++j) af[j] = naf[j];
        b0 = nb0; b1 = nb1; b2 = nb2; b3 = nb3;
      }
    }
  }
#pragma unroll
  for (int i = 0; i < 8; ++i) {
    int col = hhalf * 128 + i * 16 + lanelo;
#pragma unroll
    for (int v = 0; v < 8; ++v) {
      int M = msub * 16 + v + 8 * laneHi;
      partial[((size_t)split * N_V + m0 + M) * D_H + col] = acc[i][v];
    }
  }
}

// ------------- deterministic split-K reduce + bias + relu -------------

__global__ void k_reduce(const float* __restrict__ partial, int nsplit, size_t chunk,
                         const float* __restrict__ bias, float* __restrict__ out, int total) {
  int idx = blockIdx.x * 256 + threadIdx.x;
  if (idx >= total) return;
  float s = bias[idx & (D_H - 1)];
  for (int sp = 0; sp < nsplit; ++sp) s += partial[(size_t)sp * chunk + idx];
  out[idx] = fmaxf(s, 0.f);
}

// ---------------- launch ----------------

extern "C" void kernel_launch(void* const* d_in, const int* in_sizes, int n_in,
                              void* d_out, int out_size, void* d_ws, size_t ws_size,
                              hipStream_t stream) {
  (void)in_sizes; (void)n_in; (void)out_size; (void)ws_size;
  const float* u_feat   = (const float*)d_in[0];
  const float* v_feat   = (const float*)d_in[1];
  const int*   u        = (const int*)d_in[2];
  const int*   v        = (const int*)d_in[3];
  const float* support  = (const float*)d_in[4];
  const float* u_weight = (const float*)d_in[5];
  const float* u_bias   = (const float*)d_in[6];
  float* out = (float*)d_out;

  char* ws = (char*)d_ws;
  size_t o = 0;
  auto alloc = [&](size_t bytes) {
    size_t p = o;
    o += (bytes + 255) & ~(size_t)255;
    return p;
  };
  _Float16* wc    = (_Float16*)(ws + alloc((size_t)NUM_R * D_IN * D_H * 2));
  _Float16* tmpu  = (_Float16*)(ws + alloc((size_t)NUM_R * N_U * D_H * 2));
  _Float16* tmpv  = (_Float16*)(ws + alloc((size_t)NUM_R * N_V * D_H * 2));
  float*    cinv  = (float*)(ws + alloc((size_t)NUM_R * N_U * 4));
  float*    rinv  = (float*)(ws + alloc((size_t)NUM_R * N_V * 4));
  float*    partu = (float*)(ws + alloc((size_t)S_U * N_U * D_H * 4));
  float*    partv = (float*)(ws + alloc((size_t)S_V * N_V * D_H * 4));

  k_cumsum_w<<<(D_IN * D_H + 255) / 256, 256, 0, stream>>>(u_weight, wc);
  k_colsum<<<dim3(N_V / 256, NUM_R), 256, 0, stream>>>(support, rinv);
  k_rowsum<<<dim3(N_U, NUM_R), 256, 0, stream>>>(support, cinv);
  k_gemm_feat<<<dim3(N_U / 64, D_H / 64, NUM_R), 256, 0, stream>>>(u_feat, N_U, wc, cinv, tmpu);
  k_gemm_feat<<<dim3(N_V / 64, D_H / 64, NUM_R), 256, 0, stream>>>(v_feat, N_V, wc, rinv, tmpv);
  k_gemm_zu<<<dim3(N_U / 64, S_U), 256, 0, stream>>>(support, u, cinv, tmpv, partu);
  k_gemm_zv<<<dim3(N_V / 64, S_V), 256, 0, stream>>>(support, v, rinv, tmpu, partv);
  k_reduce<<<(N_U * D_H) / 256, 256, 0, stream>>>(partu, S_U, (size_t)N_U * D_H, u_bias,
                                                  out, N_U * D_H);
  k_reduce<<<(N_V * D_H) / 256, 256, 0, stream>>>(partv, S_V, (size_t)N_V * D_H, u_bias,
                                                  out + (size_t)N_U * D_H, N_V * D_H);
}
